// PNP_43696997270136
// MI455X (gfx1250) — matrix operand order repaired and checked by reading, not compiled, not run on hardware
//
#include <hip/hip_runtime.h>
#include <hip/hip_bf16.h>

// ---------------------------------------------------------------------------
// Types for CDNA5 WMMA (wave32): v_wmma_f32_16x16x32_bf16
// ---------------------------------------------------------------------------
typedef __attribute__((ext_vector_type(16))) __bf16 v16bf;
typedef __attribute__((ext_vector_type(8)))  float  v8f;

union Frag { unsigned int u[8]; v16bf v; };

__device__ __forceinline__ unsigned short f2bf(float f) {
    unsigned int u = __float_as_uint(f);
    u += 0x7FFFu + ((u >> 16) & 1u);            // round-to-nearest-even
    return (unsigned short)(u >> 16);
}
__device__ __forceinline__ float bf2f(unsigned short b) {
    return __uint_as_float(((unsigned int)b) << 16);
}
// order-preserving float<->uint encoding (for atomicMax on floats)
__device__ __forceinline__ unsigned int fenc(float f) {
    unsigned int u = __float_as_uint(f);
    return (u & 0x80000000u) ? ~u : (u | 0x80000000u);
}
__device__ __forceinline__ float fdec(unsigned int u) {
    return (u & 0x80000000u) ? __uint_as_float(u & 0x7FFFFFFFu)
                             : __uint_as_float(~u);
}

// ---------------------------------------------------------------------------
// Async global->LDS (GLOBAL_LOAD_ASYNC_TO_LDS_B128, ASYNCcnt-tracked).
// The builtin takes pointers to 4xi32 vectors: (global v4i*, lds v4i*, off, cpol).
// Falls back to VGPR round-trip if the builtin is not declared.
// ---------------------------------------------------------------------------
#if defined(__has_builtin)
#  if __has_builtin(__builtin_amdgcn_global_load_async_to_lds_b128)
#    define HAVE_ASYNC_LDS 1
#  endif
#endif
#ifndef HAVE_ASYNC_LDS
#  define HAVE_ASYNC_LDS 0
#endif

typedef __attribute__((ext_vector_type(4))) int v4i;
typedef __attribute__((address_space(1))) v4i gm_v4i;
typedef __attribute__((address_space(3))) v4i lds_v4i;

__device__ __forceinline__ void stage16(void* lds_dst, const void* gsrc) {
#if HAVE_ASYNC_LDS
    __builtin_amdgcn_global_load_async_to_lds_b128(
        (gm_v4i*)(void*)gsrc, (lds_v4i*)lds_dst, 0, 0);
#else
    *(uint4*)lds_dst = *(const uint4*)gsrc;
#endif
}
__device__ __forceinline__ void stage_wait() {
#if HAVE_ASYNC_LDS
    asm volatile("s_wait_asynccnt 0" ::: "memory");
#endif
}

// Problem constants
#define BATCH   4
#define NPATCH  196
#define NPAD    224          // 14 tiles of 16
#define DIM     768
#define NCLS    200
#define NVOC    500
#define CDIM    512
#define HID     1024
#define MROWS   100000       // NCLS*NVOC

// ---------------------------------------------------------------------------
__global__ void k_zero(unsigned int* __restrict__ p, int n) {
    int i = blockIdx.x * 256 + threadIdx.x;
    if (i < n) p[i] = 0u;
}

// ---------------------------------------------------------------------------
// proto = l2norm(vocab + residual) -> bf16 [MROWS][512]
// ---------------------------------------------------------------------------
__global__ __launch_bounds__(256)
void k_proto(const float* __restrict__ vocab, const float* __restrict__ res,
             unsigned short* __restrict__ proto) {
    __shared__ float red[256];
    int r = blockIdx.x, tid = threadIdx.x;
    const float* v  = vocab + (size_t)r * CDIM;
    const float* rr = res   + (size_t)r * CDIM;
    float x0 = v[tid] + rr[tid];
    float x1 = v[tid + 256] + rr[tid + 256];
    red[tid] = x0 * x0 + x1 * x1;
    __syncthreads();
    for (int s = 128; s > 0; s >>= 1) {
        if (tid < s) red[tid] += red[tid + s];
        __syncthreads();
    }
    float inv = 1.0f / fmaxf(sqrtf(red[0]), 1e-12f);
    proto[(size_t)r * CDIM + tid]       = f2bf(x0 * inv);
    proto[(size_t)r * CDIM + tid + 256] = f2bf(x1 * inv);
}

// ---------------------------------------------------------------------------
// transpose+cast: dst[n*K+k] = bf16(src[k*N+n])
// ---------------------------------------------------------------------------
__global__ void k_castT(const float* __restrict__ src,
                        unsigned short* __restrict__ dst, int K, int N) {
    int i = blockIdx.x * 256 + threadIdx.x;
    if (i >= K * N) return;
    int n = i / K, k = i % K;
    dst[i] = f2bf(src[(size_t)k * N + n]);
}

// ---------------------------------------------------------------------------
// pt = l2norm(patch_tokens) -> bf16 [B][224][768], rows >=196 zero padded
// ---------------------------------------------------------------------------
__global__ __launch_bounds__(256)
void k_pt(const float* __restrict__ tok, unsigned short* __restrict__ pt) {
    __shared__ float red[256];
    int blk = blockIdx.x;                 // 0..B*NPAD-1
    int b = blk / NPAD, n = blk % NPAD;
    int tid = threadIdx.x;
    unsigned short* dst = pt + ((size_t)b * NPAD + n) * DIM;
    if (n >= NPATCH) {
        dst[tid] = 0; dst[tid + 256] = 0; dst[tid + 512] = 0;
        return;
    }
    const float* src = tok + ((size_t)b * NPATCH + n) * DIM;
    float x0 = src[tid], x1 = src[tid + 256], x2 = src[tid + 512];
    red[tid] = x0 * x0 + x1 * x1 + x2 * x2;
    __syncthreads();
    for (int s = 128; s > 0; s >>= 1) {
        if (tid < s) red[tid] += red[tid + s];
        __syncthreads();
    }
    float inv = 1.0f / fmaxf(sqrtf(red[0]), 1e-12f);
    dst[tid] = f2bf(x0 * inv); dst[tid + 256] = f2bf(x1 * inv);
    dst[tid + 512] = f2bf(x2 * inv);
}

// ---------------------------------------------------------------------------
// saw[c][v] = softmax_v(sa[c]) * V
// ---------------------------------------------------------------------------
__global__ __launch_bounds__(512)
void k_saw(const float* __restrict__ sa, float* __restrict__ saw) {
    __shared__ float red[512];
    int c = blockIdx.x, tid = threadIdx.x;
    float x = (tid < NVOC) ? sa[c * NVOC + tid] : -3.0e38f;
    red[tid] = x;
    __syncthreads();
    for (int s = 256; s > 0; s >>= 1) {
        if (tid < s) red[tid] = fmaxf(red[tid], red[tid + s]);
        __syncthreads();
    }
    float mx = red[0];
    __syncthreads();
    float e = (tid < NVOC) ? __expf(x - mx) : 0.f;
    red[tid] = e;
    __syncthreads();
    for (int s = 256; s > 0; s >>= 1) {
        if (tid < s) red[tid] += red[tid + s];
        __syncthreads();
    }
    if (tid < NVOC) saw[c * NVOC + tid] = e * ((float)NVOC / red[0]);
}

// ---------------------------------------------------------------------------
// GEMM1: h = proto @ w1   [MROWS x 512] x [512 x 1024] -> bf16
// 64x64 block tile, BK=32; A/B staged via async global->LDS.
// ---------------------------------------------------------------------------
__global__ __launch_bounds__(256)
void k_gemm1(const unsigned short* __restrict__ A,    // proto [MROWS][512]
             const unsigned short* __restrict__ Bt,   // w1t   [1024][512]
             unsigned short* __restrict__ H) {        // h     [MROWS][1024]
    __shared__ unsigned int As32[64 * 16];   // 64 rows x 32 bf16 (k-contig)
    __shared__ unsigned int Bs32[64 * 16];   // 64 cols x 32 bf16 (k-contig)
    const int rowBase = blockIdx.x * 64, colBase = blockIdx.y * 64;
    const int tid = threadIdx.x, lane = tid & 31, wave = tid >> 5;
    const int mt = wave >> 1, ntp = (wave & 1) * 2;
    v8f acc0 = {}; v8f acc1 = {};
    int r = tid >> 2, q = tid & 3;
    int arow = rowBase + r; if (arow >= MROWS) arow = MROWS - 1;   // clamp (masked at store)
    for (int k0 = 0; k0 < CDIM; k0 += 32) {
        stage16(&As32[tid * 4], A + (size_t)arow * CDIM + k0 + q * 8);
        stage16(&Bs32[tid * 4], Bt + (size_t)(colBase + r) * CDIM + k0 + q * 8);
        stage_wait();
        __syncthreads();
        Frag a, b0, b1;
        {
            const unsigned int* Ap = As32 + (mt * 16 + (lane & 15)) * 16;
            int s = (lane >> 4) * 8;
#pragma unroll
            for (int i = 0; i < 4; i++) { a.u[i] = Ap[s + i]; a.u[4 + i] = Ap[s + 4 + i]; }
            const unsigned int* Bp0 = Bs32 + (ntp * 16 + (lane & 15)) * 16;
            const unsigned int* Bp1 = Bp0 + 16 * 16;
            int kb = (lane >> 4) * 8;
#pragma unroll
            for (int i = 0; i < 8; i++) { b0.u[i] = Bp0[kb + i]; b1.u[i] = Bp1[kb + i]; }
        }
        acc0 = __builtin_amdgcn_wmma_f32_16x16x32_bf16(false, a.v, false, b0.v,
                                                       (short)0, acc0, false, false);
        acc1 = __builtin_amdgcn_wmma_f32_16x16x32_bf16(false, a.v, false, b1.v,
                                                       (short)0, acc1, false, false);
        __syncthreads();
    }
    int n0 = colBase + ntp * 16 + (lane & 15);
#pragma unroll
    for (int rr = 0; rr < 8; rr++) {
        int grow = rowBase + mt * 16 + rr + 8 * (lane >> 4);
        if (grow < MROWS) {
            H[(size_t)grow * HID + n0]      = f2bf(acc0[rr]);
            H[(size_t)grow * HID + n0 + 16] = f2bf(acc1[rr]);
        }
    }
}

// ---------------------------------------------------------------------------
// BN column stats over h (streaming pass, block = 256 rows)
// ---------------------------------------------------------------------------
__global__ __launch_bounds__(256)
void k_colstats(const unsigned short* __restrict__ H,
                float* __restrict__ colsum, float* __restrict__ colsq) {
    int tid = threadIdx.x;
    int r0 = blockIdx.x * 256;
    int rend = r0 + 256; if (rend > MROWS) rend = MROWS;
    float s[4] = {0, 0, 0, 0}, q[4] = {0, 0, 0, 0};
    for (int r = r0; r < rend; r++) {
        const unsigned short* row = H + (size_t)r * HID;
        __builtin_prefetch(row + 8 * HID, 0, 0);   // global_prefetch_b8
#pragma unroll
        for (int j = 0; j < 4; j++) {
            float x = bf2f(row[tid + j * 256]);
            s[j] += x; q[j] += x * x;
        }
    }
#pragma unroll
    for (int j = 0; j < 4; j++) {
        atomicAdd(&colsum[tid + j * 256], s[j]);
        atomicAdd(&colsq[tid + j * 256], q[j]);
    }
}

__global__ void k_bnfin(const float* __restrict__ colsum, const float* __restrict__ colsq,
                        const float* __restrict__ gamma, const float* __restrict__ beta,
                        float* __restrict__ bna, float* __restrict__ bnb) {
    int c = threadIdx.x;                        // 1024
    const float n = (float)MROWS;
    float mu = colsum[c] / n;
    float var = colsq[c] / n - mu * mu;
    float a = rsqrtf(var + 1e-5f) * gamma[c];
    bna[c] = a;
    bnb[c] = beta[c] - mu * a;
}

// ---------------------------------------------------------------------------
// GEMM2: p = relu(h*a+b) @ w2 + b2 ; store bf16 + per-row sum(p^2).
// A staging applies BN+ReLU (VGPR path); B staged async.
// ---------------------------------------------------------------------------
__global__ __launch_bounds__(256)
void k_gemm2(const unsigned short* __restrict__ H,   // [MROWS][1024]
             const unsigned short* __restrict__ Bt,  // w2t [768][1024]
             const float* __restrict__ bna, const float* __restrict__ bnb,
             const float* __restrict__ b2,
             unsigned short* __restrict__ P,         // [MROWS][768]
             float* __restrict__ rowss) {
    __shared__ unsigned int As32[64 * 16];
    __shared__ unsigned int Bs32[64 * 16];
    __shared__ float rowacc[64];
    const int rowBase = blockIdx.x * 64, colBase = blockIdx.y * 64;
    const int tid = threadIdx.x, lane = tid & 31, wave = tid >> 5;
    const int mt = wave >> 1, ntp = (wave & 1) * 2;
    if (tid < 64) rowacc[tid] = 0.f;
    v8f acc0 = {}; v8f acc1 = {};
    int r = tid >> 2, q = tid & 3;
    for (int k0 = 0; k0 < HID; k0 += 32) {
        stage16(&Bs32[tid * 4], Bt + (size_t)(colBase + r) * HID + k0 + q * 8);
        int grow = rowBase + r;
        uint4 hv = make_uint4(0, 0, 0, 0);
        if (grow < MROWS)
            hv = *(const uint4*)(H + (size_t)grow * HID + k0 + q * 8);
        const unsigned short* hs = (const unsigned short*)&hv;
        uint4 ov; unsigned short* os = (unsigned short*)&ov;
#pragma unroll
        for (int j = 0; j < 8; j++) {
            int k = k0 + q * 8 + j;
            float y = bf2f(hs[j]) * bna[k] + bnb[k];
            os[j] = f2bf(fmaxf(y, 0.f));
        }
        ((uint4*)As32)[tid] = ov;
        stage_wait();
        __syncthreads();
        Frag a, b0, b1;
        {
            const unsigned int* Ap = As32 + (mt * 16 + (lane & 15)) * 16;
            int s = (lane >> 4) * 8;
#pragma unroll
            for (int i = 0; i < 4; i++) { a.u[i] = Ap[s + i]; a.u[4 + i] = Ap[s + 4 + i]; }
            const unsigned int* Bp0 = Bs32 + (ntp * 16 + (lane & 15)) * 16;
            const unsigned int* Bp1 = Bp0 + 16 * 16;
            int kb = (lane >> 4) * 8;
#pragma unroll
            for (int i = 0; i < 8; i++) { b0.u[i] = Bp0[kb + i]; b1.u[i] = Bp1[kb + i]; }
        }
        acc0 = __builtin_amdgcn_wmma_f32_16x16x32_bf16(false, a.v, false, b0.v,
                                                       (short)0, acc0, false, false);
        acc1 = __builtin_amdgcn_wmma_f32_16x16x32_bf16(false, a.v, false, b1.v,
                                                       (short)0, acc1, false, false);
        __syncthreads();
    }
    int n0 = colBase + ntp * 16 + (lane & 15);
    float bias0 = b2[n0], bias1 = b2[n0 + 16];
#pragma unroll
    for (int rr = 0; rr < 8; rr++) {
        int m = mt * 16 + rr + 8 * (lane >> 4);
        int grow = rowBase + m;
        if (grow < MROWS) {
            float v0 = acc0[rr] + bias0, v1 = acc1[rr] + bias1;
            P[(size_t)grow * DIM + n0]      = f2bf(v0);
            P[(size_t)grow * DIM + n0 + 16] = f2bf(v1);
            atomicAdd(&rowacc[m], v0 * v0 + v1 * v1);
        }
    }
    __syncthreads();
    if (tid < 64 && rowBase + tid < MROWS)
        atomicAdd(&rowss[rowBase + tid], rowacc[tid]);
}

// in-place l2 row normalization of P
__global__ void k_pnorm(unsigned short* __restrict__ P, const float* __restrict__ rowss) {
    size_t i = (size_t)blockIdx.x * 256 + threadIdx.x;
    if (i >= (size_t)MROWS * DIM) return;
    size_t r = i / DIM;
    float inv = 1.0f / fmaxf(sqrtf(rowss[r]), 1e-12f);
    P[i] = f2bf(bf2f(P[i]) * inv);
}

// ---------------------------------------------------------------------------
// Similarity GEMM + fused reductions, 64-column stripes.
// grid (8 v-stripes of 64, 8 c-chunks of 25, 4 batches), 224 threads (7 waves)
// Per class: WMMA 224x64 stripe (K=768); top-5 over patches -> class_logits;
// max over classes kept in an LDS buffer, committed via encoded atomicMax.
// Dynamic LDS: As 14336 | Bs 4096 | stripe 57344 | runmax 57344 = 133120 B.
// ---------------------------------------------------------------------------
#define SIM_SMEM (14336 + 4096 + 57344 + 57344)

__global__ __launch_bounds__(224)
void k_sim(const unsigned short* __restrict__ PT,   // [4][224][768]
           const unsigned short* __restrict__ PRO,  // [MROWS][768]
           const float* __restrict__ SAW,           // [200][500]
           float* __restrict__ outClass,            // [4][200]
           unsigned int* __restrict__ pmax) {       // [4][196][500] encoded
    extern __shared__ char smem[];
    unsigned int* As32 = (unsigned int*)smem;                    // 224 x 32 bf16
    unsigned int* Bs32 = (unsigned int*)(smem + 14336);          // 64  x 32 bf16
    float* stripe = (float*)(smem + 18432);                      // 224 x 64 f32
    float* runmax = (float*)(smem + 75776);                      // 224 x 64 f32
    const int b = blockIdx.z;
    const int v0 = blockIdx.x * 64;
    const int c0 = blockIdx.y * 25;
    const int tid = threadIdx.x, lane = tid & 31, wave = tid >> 5;
    const unsigned short* Abase = PT + (size_t)b * NPAD * DIM;

    for (int i = tid; i < NPAD * 64; i += 224) runmax[i] = -3.0e38f;
    __syncthreads();

    for (int ci = 0; ci < 25; ci++) {
        int c = c0 + ci;
        v8f acc[2][4];
#pragma unroll
        for (int mi = 0; mi < 2; mi++)
#pragma unroll
            for (int nt = 0; nt < 4; nt++) acc[mi][nt] = (v8f){};
        for (int k0 = 0; k0 < DIM; k0 += 32) {
#pragma unroll
            for (int it = 0; it < 4; it++) {                  // A: 896 x uint4
                int idx = tid + it * 224;
                int r = idx >> 2, q = idx & 3;
                stage16(&As32[idx * 4], Abase + (size_t)r * DIM + k0 + q * 8);
            }
            for (int idx = tid; idx < 256; idx += 224) {      // B: 256 x uint4
                int j = idx >> 2, q = idx & 3;
                int v = v0 + j; if (v >= NVOC) v = NVOC - 1;  // clamp (masked later)
                stage16(&Bs32[idx * 4], PRO + ((size_t)c * NVOC + v) * DIM + k0 + q * 8);
            }
            stage_wait();
            __syncthreads();
            Frag a0, a1, bb[4];
            {
                const unsigned int* Ap0 = As32 + ((wave * 2) * 16 + (lane & 15)) * 16;
                const unsigned int* Ap1 = Ap0 + 16 * 16;
                int s = (lane >> 4) * 8;
#pragma unroll
                for (int i = 0; i < 4; i++) {
                    a0.u[i] = Ap0[s + i]; a0.u[4 + i] = Ap0[s + 4 + i];
                    a1.u[i] = Ap1[s + i]; a1.u[4 + i] = Ap1[s + 4 + i];
                }
                int kb = (lane >> 4) * 8;
#pragma unroll
                for (int nt = 0; nt < 4; nt++) {
                    const unsigned int* Bp = Bs32 + (nt * 16 + (lane & 15)) * 16;
#pragma unroll
                    for (int i = 0; i < 8; i++) bb[nt].u[i] = Bp[kb + i];
                }
            }
#pragma unroll
            for (int nt = 0; nt < 4; nt++) {
                acc[0][nt] = __builtin_amdgcn_wmma_f32_16x16x32_bf16(
                    false, a0.v, false, bb[nt].v, (short)0, acc[0][nt], false, false);
                acc[1][nt] = __builtin_amdgcn_wmma_f32_16x16x32_bf16(
                    false, a1.v, false, bb[nt].v, (short)0, acc[1][nt], false, false);
            }
            __syncthreads();
        }
        // spill stripe to LDS
        {
            int n = lane & 15;
#pragma unroll
            for (int mi = 0; mi < 2; mi++)
#pragma unroll
                for (int nt = 0; nt < 4; nt++)
#pragma unroll
                    for (int r = 0; r < 8; r++) {
                        int m = wave * 32 + mi * 16 + r + 8 * (lane >> 4);
                        stripe[m * 64 + nt * 16 + n] = acc[mi][nt][r];
                    }
        }
        __syncthreads();
        // running max over classes (thread t owns patch-row t; staggered banks)
        {
#pragma unroll 4
            for (int jj = 0; jj < 64; jj++) {
                int j = (jj + tid) & 63;
                runmax[tid * 64 + j] = fmaxf(runmax[tid * 64 + j], stripe[tid * 64 + j]);
            }
        }
        // top-5 over patches for this (b,c,v)
        if (tid < 64 && (v0 + tid) < NVOC) {
            float t[5];
#pragma unroll
            for (int i = 0; i < 5; i++) t[i] = -3.0e38f;
            for (int n = 0; n < NPATCH; n++) {
                float val = stripe[n * 64 + tid];
                if (val > t[4]) {
                    t[4] = val;
#pragma unroll
                    for (int i = 4; i > 0; i--)
                        if (t[i] > t[i - 1]) { float tmp = t[i]; t[i] = t[i - 1]; t[i - 1] = tmp; }
                }
            }
            float mean5 = (t[0] + t[1] + t[2] + t[3] + t[4]) * 0.2f;
            atomicAdd(&outClass[b * NCLS + c], mean5 * SAW[c * NVOC + v0 + tid]);
        }
        __syncthreads();
    }
    // commit per-(b,n,v) running max across this c-chunk
    if (tid < NPATCH) {
#pragma unroll 4
        for (int jj = 0; jj < 64; jj++) {
            int j = (jj + tid) & 63;
            int v = v0 + j;
            if (v < NVOC)
                atomicMax(&pmax[((size_t)b * NPATCH + tid) * NVOC + v],
                          fenc(runmax[tid * 64 + j]));
        }
    }
}

// ---------------------------------------------------------------------------
// vocab_logits[b][v] = mean(top5_n(pmax))
// ---------------------------------------------------------------------------
__global__ void k_vocab(const unsigned int* __restrict__ pmax, float* __restrict__ outv) {
    int i = blockIdx.x * 256 + threadIdx.x;
    if (i >= BATCH * NVOC) return;
    int b = i / NVOC, v = i % NVOC;
    float t[5];
#pragma unroll
    for (int j = 0; j < 5; j++) t[j] = -3.0e38f;
    for (int n = 0; n < NPATCH; n++) {
        float val = fdec(pmax[((size_t)b * NPATCH + n) * NVOC + v]);
        if (val > t[4]) {
            t[4] = val;
#pragma unroll
            for (int j = 4; j > 0; j--)
                if (t[j] > t[j - 1]) { float tmp = t[j]; t[j] = t[j - 1]; t[j - 1] = tmp; }
        }
    }
    outv[i] = (t[0] + t[1] + t[2] + t[3] + t[4]) * 0.2f;
}

// ---------------------------------------------------------------------------
// launch
// ---------------------------------------------------------------------------
extern "C" void kernel_launch(void* const* d_in, const int* in_sizes, int n_in,
                              void* d_out, int out_size, void* d_ws, size_t ws_size,
                              hipStream_t stream) {
    const float* patch_tokens = (const float*)d_in[0];
    const float* vocab        = (const float*)d_in[1];
    const float* residual     = (const float*)d_in[2];
    const float* w1           = (const float*)d_in[3];
    const float* bn_gamma     = (const float*)d_in[4];
    const float* bn_beta      = (const float*)d_in[5];
    const float* w2           = (const float*)d_in[6];
    const float* b2           = (const float*)d_in[7];
    const float* sa_weights   = (const float*)d_in[8];
    float* out = (float*)d_out;

    char* ws = (char*)d_ws;
    const size_t OFF_PROTO = 0;
    const size_t OFF_H     = OFF_PROTO + (size_t)MROWS * CDIM * 2;
    const size_t OFF_P     = OFF_H     + (size_t)MROWS * HID  * 2;
    const size_t OFF_W1T   = OFF_P     + (size_t)MROWS * DIM  * 2;
    const size_t OFF_W2T   = OFF_W1T   + (size_t)HID * CDIM * 2;
    const size_t OFF_PT    = OFF_W2T   + (size_t)DIM * HID * 2;
    const size_t OFF_CSUM  = OFF_PT    + (size_t)BATCH * NPAD * DIM * 2;
    const size_t OFF_CSQ   = OFF_CSUM  + HID * 4;
    const size_t OFF_BNA   = OFF_CSQ   + HID * 4;
    const size_t OFF_BNB   = OFF_BNA   + HID * 4;
    const size_t OFF_RSS   = OFF_BNB   + HID * 4;
    const size_t OFF_SAW   = OFF_RSS   + (size_t)MROWS * 4;
    const size_t OFF_PMAX  = OFF_SAW   + (size_t)NCLS * NVOC * 4;

    unsigned short* proto = (unsigned short*)(ws + OFF_PROTO);
    unsigned short* h     = (unsigned short*)(ws + OFF_H);
    unsigned short* p     = (unsigned short*)(ws + OFF_P);
    unsigned short* w1t   = (unsigned short*)(ws + OFF_W1T);
    unsigned short* w2t   = (unsigned short*)(ws + OFF_W2T);
    unsigned short* pt    = (unsigned short*)(ws + OFF_PT);
    float* colsum = (float*)(ws + OFF_CSUM);
    float* colsq  = (float*)(ws + OFF_CSQ);
    float* bna    = (float*)(ws + OFF_BNA);
    float* bnb    = (float*)(ws + OFF_BNB);
    float* rowss  = (float*)(ws + OFF_RSS);
    float* saw    = (float*)(ws + OFF_SAW);
    unsigned int* pmax = (unsigned int*)(ws + OFF_PMAX);

    // init accumulators / max buffers / class output
    k_zero<<<(HID * 4 + 255) / 256, 256, 0, stream>>>((unsigned int*)colsum, HID * 4);
    k_zero<<<(MROWS + 255) / 256, 256, 0, stream>>>((unsigned int*)rowss, MROWS);
    k_zero<<<(BATCH * NPATCH * NVOC + 255) / 256, 256, 0, stream>>>(pmax, BATCH * NPATCH * NVOC);
    k_zero<<<(BATCH * NCLS + 255) / 256, 256, 0, stream>>>((unsigned int*)out, BATCH * NCLS);

    // preprocessing
    k_proto<<<MROWS, 256, 0, stream>>>(vocab, residual, proto);
    k_castT<<<(HID * CDIM + 255) / 256, 256, 0, stream>>>(w1, w1t, CDIM, HID);
    k_castT<<<(DIM * HID + 255) / 256, 256, 0, stream>>>(w2, w2t, HID, DIM);
    k_pt<<<BATCH * NPAD, 256, 0, stream>>>(patch_tokens, pt);
    k_saw<<<NCLS, 512, 0, stream>>>(sa_weights, saw);

    // prototype MLP
    k_gemm1<<<dim3((MROWS + 63) / 64, HID / 64), 256, 0, stream>>>(proto, w1t, h);
    k_colstats<<<(MROWS + 255) / 256, 256, 0, stream>>>(h, colsum, colsq);
    k_bnfin<<<1, HID, 0, stream>>>(colsum, colsq, bn_gamma, bn_beta, bna, bnb);
    k_gemm2<<<dim3((MROWS + 63) / 64, DIM / 64), 256, 0, stream>>>(h, w2t, bna, bnb, b2, p, rowss);
    k_pnorm<<<(int)(((size_t)MROWS * DIM + 255) / 256), 256, 0, stream>>>(p, rowss);

    // similarity + fused reductions
    k_sim<<<dim3((NVOC + 63) / 64, NCLS / 25, BATCH), 224, SIM_SMEM, stream>>>(
        pt, p, saw, out, pmax);
    k_vocab<<<(BATCH * NVOC + 255) / 256, 256, 0, stream>>>(pmax, out + BATCH * NCLS);
}
